// CrossAttention2D_36610301231375
// MI455X (gfx1250) — matrix operand order repaired and checked by reading, not compiled, not run on hardware
//
#include <hip/hip_runtime.h>

// ---------------------------------------------------------------------------
// CrossAttention2D fused kernel for gfx1250 (MI455X), wave32 + WMMA bf16.
// B=8, C(=TIME=SPEC)=128, F=256, H=64, W=512.
// One workgroup = one (b,h) slice x one 128-query tile. 8 waves, 256 thr.
// ---------------------------------------------------------------------------

typedef __bf16 bf16_t;
typedef bf16_t v16bf __attribute__((ext_vector_type(16)));
typedef float  v8f   __attribute__((ext_vector_type(8)));

#define B_N   8
#define C_N   128    // channel dim (TIME_DIM == SPEC_DIM == 128)
#define F_N   256    // feature dim
#define H_N   64
#define W_N   512
#define TILE  128    // query / key tile

// softmax scale (1/sqrt(256)) folded with log2(e): exp(x) == exp2(x*log2e)
#define SCL 0.09016844005556021f

__device__ __forceinline__ unsigned short f2bf(float x) {
  union { float f; unsigned int u; } v; v.f = x;
  unsigned int r = v.u + 0x7fffu + ((v.u >> 16) & 1u);   // round-to-nearest-even
  return (unsigned short)(r >> 16);
}

// build a 16-element bf16 fragment from two 16B LDS chunks (ds_load_b128 x2)
__device__ __forceinline__ v16bf ldfrag(const unsigned short* p0,
                                        const unsigned short* p1) {
  union { v16bf v; uint4 q[2]; } f;
  f.q[0] = *(const uint4*)p0;
  f.q[1] = *(const uint4*)p1;
  return f.v;
}

__device__ __forceinline__ v8f wmma_bf16(v16bf a, v16bf b, v8f c) {
  return __builtin_amdgcn_wmma_f32_16x16x32_bf16(false, a, false, b,
                                                 (short)0, c, false, false);
}

// 16-lane (row-group) reductions via ds_swizzle xor patterns (wave32)
template <int IMM>
__device__ __forceinline__ float swzf(float x) {
  return __int_as_float(__builtin_amdgcn_ds_swizzle(__float_as_int(x), IMM));
}
__device__ __forceinline__ float redmax16(float x) {
  x = fmaxf(x, swzf<0x041f>(x));   // xor 1
  x = fmaxf(x, swzf<0x081f>(x));   // xor 2
  x = fmaxf(x, swzf<0x101f>(x));   // xor 4
  x = fmaxf(x, swzf<0x201f>(x));   // xor 8
  return x;
}
__device__ __forceinline__ float redsum16(float x) {
  x += swzf<0x041f>(x);
  x += swzf<0x081f>(x);
  x += swzf<0x101f>(x);
  x += swzf<0x201f>(x);
  return x;
}

// Project one 16-row strip: out[16 x 256] = X[16 x 128] * W^T + bias,
// X in lXp[w][c] (A-matrix), W in lWp[f][c] (B-matrix rows = N cols, contiguous K)
__device__ __forceinline__ void proj_strip(const unsigned short* lXp,
                                           const unsigned short* lWp,
                                           const float* __restrict__ bias,
                                           unsigned short* lOut,
                                           int wave, int ln, int hv) {
  const unsigned short* aRow = lXp + (wave * 16 + ln) * C_N;
#pragma unroll
  for (int nt = 0; nt < 16; ++nt) {
    const int f = nt * 16 + ln;
    const float bv = bias[f];
    v8f acc = { bv, bv, bv, bv, bv, bv, bv, bv };
    const unsigned short* bRow = lWp + f * C_N;
#pragma unroll
    for (int kk = 0; kk < 4; ++kk) {          // K = 128 in steps of 32
      v16bf A  = ldfrag(aRow + kk * 32 + 8 * hv,
                        aRow + kk * 32 + 16 + 8 * hv);
      v16bf Bm = ldfrag(bRow + kk * 32 + 8 * hv,
                        bRow + kk * 32 + 16 + 8 * hv);
      acc = wmma_bf16(A, Bm, acc);
    }
#pragma unroll
    for (int v = 0; v < 8; ++v)
      lOut[(wave * 16 + v + 8 * hv) * F_N + f] = f2bf(acc[v]);
  }
}

__global__ __launch_bounds__(256, 1)
void ca2d_fused_kernel(const float* __restrict__ wav,
                       const float* __restrict__ spec,
                       const float* __restrict__ w1, const float* __restrict__ b1,
                       const float* __restrict__ w2, const float* __restrict__ b2,
                       float* __restrict__ out) {
  // 288 KB LDS total (fits CDNA5's 320 KB/WGP)
  __shared__ __align__(16) unsigned short lQ[TILE * F_N];   // 64 KB  Q tile bf16
  __shared__ __align__(16) unsigned short lK[TILE * F_N];   // 64 KB  K tile bf16
  __shared__ __align__(16) unsigned short lW[F_N * C_N];    // 64 KB  W1 then W2
  __shared__ __align__(16) unsigned short lV[C_N * TILE];   // 32 KB  V^T = spec tile
  __shared__ __align__(16) unsigned short lX[TILE * C_N];   // 32 KB  transposed input
  __shared__ __align__(16) unsigned short lP[TILE * TILE];  // 32 KB  P bf16

  const int t    = threadIdx.x;
  const int wave = t >> 5;
  const int lane = t & 31;
  const int hv   = lane >> 4;       // half-wave select
  const int ln   = lane & 15;

  const int bh = blockIdx.x >> 2;   // 0..511 : (b,h)
  const int qt = blockIdx.x & 3;    // query tile
  const int b  = bh >> 6;
  const int h  = bh & 63;
  const int q0 = qt * TILE;

  // ---- stage W1 (bf16) ------------------------------------------------------
  for (int i = t; i < F_N * C_N; i += 256) lW[i] = f2bf(w1[i]);

  // ---- stage waveform tile transposed: lX[w][c] -----------------------------
  {
    const int c  = t >> 1;
    const int wh = (t & 1) * 64;
    const float* src = wav + (((size_t)b * C_N + c) * H_N + h) * W_N + q0 + wh;
#pragma unroll 4
    for (int j = 0; j < 64; ++j)
      lX[(wh + j) * C_N + c] = f2bf(src[j]);
  }
  __syncthreads();

  // ---- Q projection (each wave writes its own 16-row strip of lQ) -----------
  proj_strip(lX, lW, b1, lQ, wave, ln, hv);
  __syncthreads();

  // ---- stage W2 (bf16) ------------------------------------------------------
  for (int i = t; i < F_N * C_N; i += 256) lW[i] = f2bf(w2[i]);

  // ---- flash-attention state (rows v + 8*hv of this wave's strip) -----------
  float m[8], l[8], o[8][8];
#pragma unroll
  for (int v = 0; v < 8; ++v) { m[v] = -1e30f; l[v] = 0.f; }
#pragma unroll
  for (int nt = 0; nt < 8; ++nt)
#pragma unroll
    for (int v = 0; v < 8; ++v) o[nt][v] = 0.f;

  for (int kt = 0; kt < 4; ++kt) {
    const int k0 = kt * TILE;
    __syncthreads();   // all waves done with lK/lV/lX (and W2 staged, iter 0)

    // -- stage spectrogram tile: lV[c][k] straight + lX[k][c] transposed ------
    {
      const int c  = t >> 1;
      const int wh = (t & 1) * 64;
      const float* src = spec + (((size_t)b * C_N + c) * H_N + h) * W_N + k0 + wh;
      unsigned int* vrow = (unsigned int*)(lV + c * TILE + wh);
#pragma unroll 4
      for (int j = 0; j < 64; j += 2) {
        float x0 = src[j], x1 = src[j + 1];
        unsigned short h0 = f2bf(x0), h1 = f2bf(x1);
        vrow[j >> 1] = (unsigned int)h0 | ((unsigned int)h1 << 16);
        lX[(wh + j)     * C_N + c] = h0;
        lX[(wh + j + 1) * C_N + c] = h1;
      }
    }
    __syncthreads();

    // -- K projection into lK (wave-private strip writes) ---------------------
    proj_strip(lX, lW, b2, lK, wave, ln, hv);
    __syncthreads();   // lK consumed by all waves below

    // -- S = Q K^T for this wave's 16-query strip (scaled, log2 domain) -------
    float p[8][8];
#pragma unroll
    for (int nt = 0; nt < 8; ++nt) {
      v8f acc = { 0.f, 0.f, 0.f, 0.f, 0.f, 0.f, 0.f, 0.f };
      const unsigned short* aRow = lQ + (wave * 16 + ln) * F_N;
      const unsigned short* bRow = lK + (nt * 16 + ln) * F_N;
#pragma unroll
      for (int kk = 0; kk < 8; ++kk) {        // K = 256 in steps of 32
        v16bf A  = ldfrag(aRow + kk * 32 + 8 * hv,
                          aRow + kk * 32 + 16 + 8 * hv);
        v16bf Bm = ldfrag(bRow + kk * 32 + 8 * hv,
                          bRow + kk * 32 + 16 + 8 * hv);
        acc = wmma_bf16(A, Bm, acc);
      }
#pragma unroll
      for (int v = 0; v < 8; ++v) p[nt][v] = acc[v] * SCL;
    }

    // -- online softmax (rows live across 16 lanes of a half-wave) ------------
#pragma unroll
    for (int v = 0; v < 8; ++v) {
      float mt = p[0][v];
#pragma unroll
      for (int nt = 1; nt < 8; ++nt) mt = fmaxf(mt, p[nt][v]);
      mt = redmax16(mt);
      const float mn    = fmaxf(m[v], mt);
      const float alpha = exp2f(m[v] - mn);
      float rs = 0.f;
#pragma unroll
      for (int nt = 0; nt < 8; ++nt) {
        float e = exp2f(p[nt][v] - mn);
        p[nt][v] = e;
        rs += e;
      }
      rs = redsum16(rs);
      m[v] = mn;
      l[v] = l[v] * alpha + rs;
#pragma unroll
      for (int nt = 0; nt < 8; ++nt) o[nt][v] *= alpha;
    }

    // -- P -> bf16 in wave-private strip of lP (same-wave DS is in-order) -----
#pragma unroll
    for (int nt = 0; nt < 8; ++nt)
#pragma unroll
      for (int v = 0; v < 8; ++v)
        lP[(wave * 16 + v + 8 * hv) * TILE + nt * 16 + ln] = f2bf(p[nt][v]);

    // -- O += P * V   (V^T is lV[c][k]: B-fragments read contiguous k) --------
    {
      const unsigned short* aRow = lP + (wave * 16 + ln) * TILE;
#pragma unroll
      for (int nt = 0; nt < 8; ++nt) {        // output-channel tile
        v8f acc = { o[nt][0], o[nt][1], o[nt][2], o[nt][3],
                    o[nt][4], o[nt][5], o[nt][6], o[nt][7] };
        const unsigned short* bRow = lV + (nt * 16 + ln) * TILE;
#pragma unroll
        for (int kk = 0; kk < 4; ++kk) {      // K = 128 keys in steps of 32
          v16bf A  = ldfrag(aRow + kk * 32 + 8 * hv,
                            aRow + kk * 32 + 16 + 8 * hv);
          v16bf Bm = ldfrag(bRow + kk * 32 + 8 * hv,
                            bRow + kk * 32 + 16 + 8 * hv);
          acc = wmma_bf16(A, Bm, acc);
        }
#pragma unroll
        for (int v = 0; v < 8; ++v) o[nt][v] = acc[v];
      }
    }
  }

  // ---- epilogue: normalize and scatter to out[b][c][h][w] (fp32) ------------
#pragma unroll
  for (int v = 0; v < 8; ++v) l[v] = 1.0f / l[v];
#pragma unroll
  for (int nt = 0; nt < 8; ++nt) {
    const int c = nt * 16 + ln;
    float* dst = out + (((size_t)b * C_N + c) * H_N + h) * W_N + q0 + wave * 16;
#pragma unroll
    for (int v = 0; v < 8; ++v)
      dst[v + 8 * hv] = o[nt][v] * l[v];
  }
}

extern "C" void kernel_launch(void* const* d_in, const int* in_sizes, int n_in,
                              void* d_out, int out_size, void* d_ws, size_t ws_size,
                              hipStream_t stream) {
  (void)in_sizes; (void)n_in; (void)d_ws; (void)ws_size; (void)out_size;
  const float* wav  = (const float*)d_in[0];
  const float* spec = (const float*)d_in[1];
  const float* w1   = (const float*)d_in[2];
  const float* b1   = (const float*)d_in[3];
  const float* w2   = (const float*)d_in[4];
  const float* b2   = (const float*)d_in[5];
  float* out = (float*)d_out;

  dim3 grid(B_N * H_N * (W_N / TILE));   // 2048 workgroups
  dim3 block(256);                        // 8 waves (wave32)
  hipLaunchKernelGGL(ca2d_fused_kernel, grid, block, 0, stream,
                     wav, spec, w1, b1, w2, b2, out);
}
